// GCNEdgeBasedEdgeGenCluster_32701880992047
// MI455X (gfx1250) — matrix operand order repaired and checked by reading, not compiled, not run on hardware
//
#include <hip/hip_runtime.h>
#include <math.h>

typedef __attribute__((ext_vector_type(16))) _Float16 v16h;
typedef __attribute__((ext_vector_type(8)))  _Float16 v8h;
typedef __attribute__((ext_vector_type(8)))  float    v8f;

#define N_NODES 100000
#define N_EDGES 2400000
#define HID 32
#define NCLUST 30

// ---------------------------------------------------------------------------
// WMMA helper: D = A(16x32 f16) * B(32x16 f16) + C(f32)
// ---------------------------------------------------------------------------
__device__ __forceinline__ v8f wmma_f16(v16h a, v16h b, v8f c) {
  return __builtin_amdgcn_wmma_f32_16x16x32_f16(
      /*neg_a=*/false, a, /*neg_b=*/false, b,
      /*c_mod=*/(short)0, c, /*reuse_a=*/false, /*reuse_b=*/false);
}

__device__ __forceinline__ void atomic_add_f32(float* p, float v) {
  unsafeAtomicAdd(p, v);  // lowers to global_atomic_add_f32 on gfx1250
}

// A-matrix lane load from a row-major 32-half row.
// Lane L (0..15): row M=L, K = {kb..kb+7, kb+16..kb+23} with kb=0.
// Lane L (16..31): same row index (L&15), kb=8.
__device__ __forceinline__ v16h load_a_f16(const _Float16* __restrict__ row, int kb) {
  v8h lo = *(const v8h*)(row + kb);
  v8h hi = *(const v8h*)(row + kb + 16);
  v16h a;
#pragma unroll
  for (int i = 0; i < 8; ++i) { a[i] = lo[i]; a[8 + i] = hi[i]; }
  return a;
}

// B panel: prepacked as [32 lanes][16 halfs] contiguous.
__device__ __forceinline__ v16h load_b_panel(const _Float16* __restrict__ panel, int lane) {
  return *(const v16h*)(panel + lane * 16);
}

// ---------------------------------------------------------------------------
// Utility kernels
// ---------------------------------------------------------------------------
__global__ void zero_f32(float* __restrict__ p, int n) {
  int i = blockIdx.x * blockDim.x + threadIdx.x;
  int stride = gridDim.x * blockDim.x;
  for (; i < n; i += stride) p[i] = 0.0f;
}

// Pack fp32 weight W[out_rows][ld] into a f16 WMMA B-panel for output columns
// [n_base, n_base+16) over K-chunk [koff, koff+32).
// B layout (16-bit, 32x16): lane L holds column n=L&15, K = {kb..kb+7, kb+16..kb+23},
// kb = (L<16)?0:8, two halfs per VGPR => per-lane data is
// W[n][koff+kb .. koff+kb+7] then W[n][koff+kb+16 .. koff+kb+23].
__global__ void pack_panel(const float* __restrict__ W, int ld, int koff,
                           int n_base, int out_rows, _Float16* __restrict__ dst) {
  int lane = threadIdx.x & 31;
  int n = n_base + (lane & 15);
  int kb = (lane < 16) ? 0 : 8;
  _Float16* d = dst + lane * 16;
#pragma unroll
  for (int t = 0; t < 16; ++t) {
    float v = (n < out_rows) ? W[n * ld + koff + kb + (t < 8 ? t : t + 8)] : 0.0f;
    d[t] = (_Float16)v;
  }
}

// segment_sum of raw fp32 edge features into agg[src[e]][c] (one elem/thread)
__global__ void scatter_edge_vals(const float* __restrict__ ev,
                                  const int* __restrict__ src,
                                  float* __restrict__ agg) {
  int i = blockIdx.x * blockDim.x + threadIdx.x;  // < N_EDGES*32
  int e = i >> 5;
  int c = i & 31;
  atomic_add_f32(&agg[src[e] * HID + c], ev[i]);
}

// ---------------------------------------------------------------------------
// Node convolution: Xout = relu( (agg/D) @ pw^T + Xin @ sw^T + pb + sb )
// One wave per 16-row tile. 16x32 output per wave (2 WMMA column-tiles).
// ---------------------------------------------------------------------------
template <bool HAS_X>
__global__ void node_conv(const float* __restrict__ agg,
                          const _Float16* __restrict__ Xin,
                          const float* __restrict__ Dv,
                          const _Float16* __restrict__ panels, int pw_idx, int sw_idx,
                          const float* __restrict__ pb, const float* __restrict__ sb,
                          _Float16* __restrict__ Xout, int ntiles) {
  int lane = threadIdx.x & 31;
  int wid = blockIdx.x * (blockDim.x >> 5) + (threadIdx.x >> 5);
  if (wid >= ntiles) return;
  int row0 = wid * 16;
  int r = row0 + (lane & 15);
  int kb = (lane < 16) ? 0 : 8;
  float inv = 1.0f / Dv[r];

  const float* ap = agg + r * HID;
  v8f alo = *(const v8f*)(ap + kb);
  v8f ahi = *(const v8f*)(ap + kb + 16);
  v16h A;
#pragma unroll
  for (int i = 0; i < 8; ++i) {
    A[i]     = (_Float16)(alo[i] * inv);
    A[8 + i] = (_Float16)(ahi[i] * inv);
  }
  v16h Ax;
  if (HAS_X) Ax = load_a_f16(Xin + r * HID, kb);

  int hi = lane >> 4;
#pragma unroll
  for (int t = 0; t < 2; ++t) {
    v8f acc = {};
    acc = wmma_f16(A, load_b_panel(panels + (pw_idx + t) * 512, lane), acc);
    if (HAS_X)
      acc = wmma_f16(Ax, load_b_panel(panels + (sw_idx + t) * 512, lane), acc);
    int n = (lane & 15) + 16 * t;
    float bias = pb[n] + sb[n];
#pragma unroll
    for (int rr = 0; rr < 8; ++rr) {
      float v = acc[rr] + bias;
      v = v > 0.0f ? v : 0.0f;
      Xout[(row0 + rr + 8 * hi) * HID + n] = (_Float16)v;
    }
  }
}

// ---------------------------------------------------------------------------
// Edge convolution: ev' = relu( E @ pw^T + ev @ sw^T + pb + sb )
//   E = [ (x1-x2)*0.5 | (x1+x2)*0.5 ]   (K = 64 -> two WMMAs)
// Fuses the downstream segment_sum: atomic-add each output row into
// agg_next[src[e]]. One wave per 16-edge tile; 6 WMMAs per wave.
// ---------------------------------------------------------------------------
template <bool EVF32, bool STORE_EV>
__global__ void edge_conv(const _Float16* __restrict__ Xh,
                          const float* __restrict__ ev32,
                          const _Float16* __restrict__ ev16,
                          const int* __restrict__ src, const int* __restrict__ dst,
                          const _Float16* __restrict__ panels, int pw_idx, int sw_idx,
                          const float* __restrict__ pb, const float* __restrict__ sb,
                          _Float16* __restrict__ evout, float* __restrict__ aggout) {
  int lane = threadIdx.x & 31;
  int wid = blockIdx.x * (blockDim.x >> 5) + (threadIdx.x >> 5);
  int e0 = wid * 16;
  int e = e0 + (lane & 15);
  int kb = (lane < 16) ? 0 : 8;

  int s = src[e], d = dst[e];
  const _Float16* r1 = Xh + s * HID;
  const _Float16* r2 = Xh + d * HID;
  v8h a1 = *(const v8h*)(r1 + kb), b1 = *(const v8h*)(r1 + kb + 16);
  v8h a2 = *(const v8h*)(r2 + kb), b2 = *(const v8h*)(r2 + kb + 16);

  const _Float16 h = (_Float16)0.5f;
  v16h Ad, As;
#pragma unroll
  for (int i = 0; i < 8; ++i) {
    Ad[i]     = (_Float16)((a1[i] - a2[i]) * h);
    As[i]     = (_Float16)((a1[i] + a2[i]) * h);
    Ad[8 + i] = (_Float16)((b1[i] - b2[i]) * h);
    As[8 + i] = (_Float16)((b1[i] + b2[i]) * h);
  }

  v16h Ae;
  if (EVF32) {
    const float* ep = ev32 + e * HID;
    v8f lo = *(const v8f*)(ep + kb);
    v8f hi2 = *(const v8f*)(ep + kb + 16);
#pragma unroll
    for (int i = 0; i < 8; ++i) { Ae[i] = (_Float16)lo[i]; Ae[8 + i] = (_Float16)hi2[i]; }
  } else {
    Ae = load_a_f16(ev16 + e * HID, kb);
  }

  int hi = lane >> 4;
  int srow[8];
#pragma unroll
  for (int rr = 0; rr < 8; ++rr) srow[rr] = src[e0 + rr + 8 * hi];

#pragma unroll
  for (int t = 0; t < 2; ++t) {
    v8f acc = {};
    acc = wmma_f16(Ad, load_b_panel(panels + (pw_idx + t) * 512, lane), acc);      // K 0..31 (diff)
    acc = wmma_f16(As, load_b_panel(panels + (pw_idx + 2 + t) * 512, lane), acc);  // K 32..63 (sum)
    acc = wmma_f16(Ae, load_b_panel(panels + (sw_idx + t) * 512, lane), acc);      // ev term
    int n = (lane & 15) + 16 * t;
    float bias = pb[n] + sb[n];
#pragma unroll
    for (int rr = 0; rr < 8; ++rr) {
      float v = acc[rr] + bias;
      v = v > 0.0f ? v : 0.0f;
      if (STORE_EV) evout[(e0 + rr + 8 * hi) * HID + n] = (_Float16)v;
      atomic_add_f32(&aggout[srow[rr] * HID + n], v);  // fused segment_sum
    }
  }
}

// ---------------------------------------------------------------------------
// Final layer: FX = (agg/D) @ n3_pw^T + Xh @ n3_sw^T + bias  (30 cols, padded
// to 32 with zero weights), then row softmax via per-wave LDS transpose.
// ---------------------------------------------------------------------------
__global__ void node_conv_final(const float* __restrict__ agg,
                                const _Float16* __restrict__ Xin,
                                const float* __restrict__ Dv,
                                const _Float16* __restrict__ panels, int pw_idx, int sw_idx,
                                const float* __restrict__ pb, const float* __restrict__ sb,
                                float* __restrict__ out, int ntiles) {
  __shared__ float lds[8][16][32];
  int lane = threadIdx.x & 31;
  int w = threadIdx.x >> 5;
  int wid = blockIdx.x * (blockDim.x >> 5) + w;
  if (wid >= ntiles) return;
  int row0 = wid * 16;
  int r = row0 + (lane & 15);
  int kb = (lane < 16) ? 0 : 8;
  float inv = 1.0f / Dv[r];

  const float* ap = agg + r * HID;
  v8f alo = *(const v8f*)(ap + kb);
  v8f ahi = *(const v8f*)(ap + kb + 16);
  v16h A;
#pragma unroll
  for (int i = 0; i < 8; ++i) {
    A[i]     = (_Float16)(alo[i] * inv);
    A[8 + i] = (_Float16)(ahi[i] * inv);
  }
  v16h Ax = load_a_f16(Xin + r * HID, kb);

  int hi = lane >> 4;
#pragma unroll
  for (int t = 0; t < 2; ++t) {
    v8f acc = {};
    acc = wmma_f16(A, load_b_panel(panels + (pw_idx + t) * 512, lane), acc);
    acc = wmma_f16(Ax, load_b_panel(panels + (sw_idx + t) * 512, lane), acc);
    int n = (lane & 15) + 16 * t;
    float bias = (n < NCLUST) ? (pb[n] + sb[n]) : 0.0f;
#pragma unroll
    for (int rr = 0; rr < 8; ++rr)
      lds[w][rr + 8 * hi][n] = acc[rr] + bias;
  }

  // in-wave LDS transpose handoff: order + wait on DScnt
  __builtin_amdgcn_wave_barrier();
  asm volatile("s_wait_dscnt 0" ::: "memory");

  if (lane < 16) {
    float* rowv = lds[w][lane];
    float mx = rowv[0];
#pragma unroll
    for (int j = 1; j < NCLUST; ++j) mx = fmaxf(mx, rowv[j]);
    float ssum = 0.0f;
#pragma unroll
    for (int j = 0; j < NCLUST; ++j) {
      float ex = __expf(rowv[j] - mx);
      rowv[j] = ex;
      ssum += ex;
    }
    float is = 1.0f / ssum;
    float* op = out + (row0 + lane) * NCLUST;
#pragma unroll
    for (int j = 0; j < NCLUST; ++j) op[j] = rowv[j] * is;
  }
}

// ---------------------------------------------------------------------------
// Host launcher
// ---------------------------------------------------------------------------
extern "C" void kernel_launch(void* const* d_in, const int* in_sizes, int n_in,
                              void* d_out, int out_size, void* d_ws, size_t ws_size,
                              hipStream_t stream) {
  (void)in_sizes; (void)n_in; (void)out_size; (void)ws_size;
  const float* Dv    = (const float*)d_in[1];
  const int*   src   = (const int*)d_in[2];
  const int*   dst   = (const int*)d_in[3];
  const float* ev0   = (const float*)d_in[4];
  const float* n1_pw = (const float*)d_in[5],  *n1_pb = (const float*)d_in[6];
  const float* n1_sb = (const float*)d_in[8];
  const float* e1_pw = (const float*)d_in[9],  *e1_pb = (const float*)d_in[10];
  const float* e1_sw = (const float*)d_in[11], *e1_sb = (const float*)d_in[12];
  const float* n2_pw = (const float*)d_in[13], *n2_pb = (const float*)d_in[14];
  const float* n2_sw = (const float*)d_in[15], *n2_sb = (const float*)d_in[16];
  const float* e2_pw = (const float*)d_in[17], *e2_pb = (const float*)d_in[18];
  const float* e2_sw = (const float*)d_in[19], *e2_sb = (const float*)d_in[20];
  const float* n3_pw = (const float*)d_in[21], *n3_pb = (const float*)d_in[22];
  const float* n3_sw = (const float*)d_in[23], *n3_sb = (const float*)d_in[24];

  // workspace layout
  char* ws = (char*)d_ws;
  size_t off = 0;
  auto alloc = [&](size_t bytes) -> void* {
    void* p = ws + off;
    off += (bytes + 255) & ~(size_t)255;
    return p;
  };
  _Float16* panels = (_Float16*)alloc(22 * 512 * sizeof(_Float16));
  float* agg  = (float*)alloc((size_t)3 * N_NODES * HID * sizeof(float));
  float* agg1 = agg;
  float* agg2 = agg + (size_t)N_NODES * HID;
  float* agg3 = agg + (size_t)2 * N_NODES * HID;
  _Float16* Xh1 = (_Float16*)alloc((size_t)N_NODES * HID * sizeof(_Float16));
  _Float16* Xh2 = (_Float16*)alloc((size_t)N_NODES * HID * sizeof(_Float16));
  _Float16* ev1 = (_Float16*)alloc((size_t)N_EDGES * HID * sizeof(_Float16));

  // 1) zero aggregation buffers (harness does not re-zero between replays)
  zero_f32<<<2048, 256, 0, stream>>>(agg, 3 * N_NODES * HID);

  // 2) pack f16 WMMA B-panels
  auto pack = [&](const float* W, int ld, int koff, int nb, int orr, int idx) {
    pack_panel<<<1, 32, 0, stream>>>(W, ld, koff, nb, orr, panels + idx * 512);
  };
  pack(n1_pw, 32, 0, 0, 32, 0);  pack(n1_pw, 32, 0, 16, 32, 1);
  pack(n2_pw, 32, 0, 0, 32, 2);  pack(n2_pw, 32, 0, 16, 32, 3);
  pack(n2_sw, 32, 0, 0, 32, 4);  pack(n2_sw, 32, 0, 16, 32, 5);
  pack(n3_pw, 32, 0, 0, 30, 6);  pack(n3_pw, 32, 0, 16, 30, 7);
  pack(n3_sw, 32, 0, 0, 30, 8);  pack(n3_sw, 32, 0, 16, 30, 9);
  pack(e1_pw, 64, 0, 0, 32, 10); pack(e1_pw, 64, 0, 16, 32, 11);
  pack(e1_pw, 64, 32, 0, 32, 12); pack(e1_pw, 64, 32, 16, 32, 13);
  pack(e1_sw, 32, 0, 0, 32, 14); pack(e1_sw, 32, 0, 16, 32, 15);
  pack(e2_pw, 64, 0, 0, 32, 16); pack(e2_pw, 64, 0, 16, 32, 17);
  pack(e2_pw, 64, 32, 0, 32, 18); pack(e2_pw, 64, 32, 16, 32, 19);
  pack(e2_sw, 32, 0, 0, 32, 20); pack(e2_sw, 32, 0, 16, 32, 21);

  // 3) segment_sum(edge_vals, src) -> agg1
  scatter_edge_vals<<<(N_EDGES * 32) / 256, 256, 0, stream>>>(ev0, src, agg1);

  const int ntiles_n  = N_NODES / 16;              // 6250
  const int nblocks_n = (ntiles_n + 7) / 8;        // 782 (8 waves/block)
  const int ntiles_e  = N_EDGES / 16;              // 150000
  const int nblocks_e = ntiles_e / 8;              // 18750

  // 4) node conv 1 (X-term is zero -> bias only)
  node_conv<false><<<nblocks_n, 256, 0, stream>>>(
      agg1, nullptr, Dv, panels, 0, 0, n1_pb, n1_sb, Xh1, ntiles_n);

  // 5) edge conv 1 (+ fused scatter into agg2)
  edge_conv<true, true><<<nblocks_e, 256, 0, stream>>>(
      Xh1, ev0, nullptr, src, dst, panels, 10, 14, e1_pb, e1_sb, ev1, agg2);

  // 6) node conv 2
  node_conv<true><<<nblocks_n, 256, 0, stream>>>(
      agg2, Xh1, Dv, panels, 2, 4, n2_pb, n2_sb, Xh2, ntiles_n);

  // 7) edge conv 2 (+ fused scatter into agg3; ev2 not needed downstream)
  edge_conv<false, false><<<nblocks_e, 256, 0, stream>>>(
      Xh2, nullptr, ev1, src, dst, panels, 16, 20, e2_pb, e2_sb, nullptr, agg3);

  // 8) final node conv + softmax
  node_conv_final<<<nblocks_n, 256, 0, stream>>>(
      agg3, Xh2, Dv, panels, 6, 8, n3_pb, n3_sb, (float*)d_out, ntiles_n);
}
